// MultiHeadAttention_65695819759720
// MI455X (gfx1250) — matrix operand order repaired and checked by reading, not compiled, not run on hardware
//
#include <hip/hip_runtime.h>
#include <hip/hip_bf16.h>

// ---------------------------------------------------------------------------
// Types for CDNA5 WMMA (wave32): v16bf = 16 bf16 (8 VGPRs), v8f = 8 f32.
// ---------------------------------------------------------------------------
typedef __attribute__((ext_vector_type(16))) __bf16 v16bf;
typedef __attribute__((ext_vector_type(8)))  __bf16 v8bf;
typedef __attribute__((ext_vector_type(8)))  float  v8f;
typedef __attribute__((ext_vector_type(4)))  unsigned int u32x4;
typedef __attribute__((ext_vector_type(8)))  int   i32x8;
typedef __attribute__((ext_vector_type(4)))  int   i32x4;

union V16 {
    v16bf v;
    v8bf  h[2];
};

#if __has_builtin(__builtin_amdgcn_tensor_load_to_lds) && \
    __has_builtin(__builtin_amdgcn_s_wait_tensorcnt)
#define USE_TDM 1
#else
#define USE_TDM 0
#endif

__device__ __forceinline__ v8f wmma_bf16(v16bf a, v16bf b, v8f c) {
    // D = A(16x32 bf16) * B(32x16 bf16) + C(16x16 f32)
    return __builtin_amdgcn_wmma_f32_16x16x32_bf16(
        /*neg_a=*/false, a, /*neg_b=*/false, b,
        /*c_mod=*/(short)0, c, /*reuse_a=*/false, /*reuse_b=*/false);
}

__device__ __forceinline__ unsigned short f2bf(float f) {
    unsigned int u = __builtin_bit_cast(unsigned int, f);
    u += 0x7fffu + ((u >> 16) & 1u);   // round-to-nearest-even
    return (unsigned short)(u >> 16);
}

__device__ __forceinline__ float rowmax16(float x) {
    x = fmaxf(x, __shfl_xor(x, 1, 32));
    x = fmaxf(x, __shfl_xor(x, 2, 32));
    x = fmaxf(x, __shfl_xor(x, 4, 32));
    x = fmaxf(x, __shfl_xor(x, 8, 32));
    return x;
}
__device__ __forceinline__ float rowsum16(float x) {
    x += __shfl_xor(x, 1, 32);
    x += __shfl_xor(x, 2, 32);
    x += __shfl_xor(x, 4, 32);
    x += __shfl_xor(x, 8, 32);
    return x;
}

// ---------------------------------------------------------------------------
// TDM: DMA a 64x64 bf16 tile (row stride = strideElems) from global into LDS.
// Packs a Tensor DMA Descriptor (D#): group0 = count/lds_addr/global_addr/type,
// group1 = data_size(2B) + tensor/tile dims + 48-bit dim0 stride.
// ---------------------------------------------------------------------------
__device__ __forceinline__ void tile_load_64x64(unsigned short* ldsDst,
                                                const unsigned short* gsrc,
                                                unsigned strideElems,
                                                int lane) {
#if USE_TDM
    const unsigned ldsOff = (unsigned)(unsigned long long)ldsDst; // LDS offset
    const unsigned long long ga = (unsigned long long)gsrc;       // byte addr
    u32x4 g0 = { 1u,                                   // count=1 (user mode)
                 ldsOff,                               // lds_addr
                 (unsigned)ga,                         // global_addr[31:0]
                 ((unsigned)(ga >> 32) & 0x01ffffffu)  // global_addr[56:32]
                     | (2u << 30) };                   // type = 2 ("image")
    i32x8 g1;
    g1[0] = (int)(1u << 16);          // wg_mask=0, data_size=1 (2 bytes)
    g1[1] = (int)(64u << 16);         // tensor_dim0[15:0] = 64
    g1[2] = (int)(64u << 16);         // tensor_dim0 hi = 0 | tensor_dim1 = 64
    g1[3] = (int)(64u << 16);         // tensor_dim1 hi = 0 | tile_dim0 = 64
    g1[4] = 64;                       // tile_dim1 = 64, tile_dim2 = 0
    g1[5] = (int)strideElems;         // tensor_dim0_stride[31:0]
    g1[6] = 0;                        // stride0 hi | tensor_dim1_stride lo
    g1[7] = 0;
    i32x4 z4 = {0, 0, 0, 0};
#if defined(__clang_major__) && __clang_major__ >= 23
    i32x8 z8 = {0, 0, 0, 0, 0, 0, 0, 0};
    __builtin_amdgcn_tensor_load_to_lds(g0, g1, z4, z4, z8, 0);
#else
    __builtin_amdgcn_tensor_load_to_lds(g0, g1, z4, z4, 0);
#endif
#else
    // Fallback: vectorized per-lane copy (64x64 shorts = 512 x 8-short chunks)
#pragma unroll
    for (int i = 0; i < 16; ++i) {
        const int chunk = lane + 32 * i;
        const int r = chunk >> 3;
        const int c = (chunk & 7) * 8;
        *(v8bf*)(ldsDst + r * 64 + c) =
            *(const v8bf*)(gsrc + (size_t)r * strideElems + c);
    }
#endif
}

__device__ __forceinline__ void tdm_wait_le2() {
#if USE_TDM
    __builtin_amdgcn_s_wait_tensorcnt(2);
#else
    __syncthreads();
#endif
}
__device__ __forceinline__ void tdm_wait_0() {
#if USE_TDM
    __builtin_amdgcn_s_wait_tensorcnt(0);
#else
    __syncthreads();
#endif
}

// ---------------------------------------------------------------------------
// Elementwise f32 -> bf16 conversion
// ---------------------------------------------------------------------------
__global__ void f32_to_bf16_kernel(const float* __restrict__ in,
                                   unsigned short* __restrict__ out,
                                   unsigned long long n) {
    unsigned long long i =
        (unsigned long long)blockIdx.x * blockDim.x + threadIdx.x;
    if (i < n) out[i] = f2bf(in[i]);
}

// ---------------------------------------------------------------------------
// Transpose V slice of qkv into vT[b, h, dh, T] (bf16): makes the K-dim
// (sequence) of the P@V B-operand contiguous.
// ---------------------------------------------------------------------------
__global__ void transpose_v_kernel(const unsigned short* __restrict__ qkv,
                                   unsigned short* __restrict__ vT) {
    unsigned long long i =
        (unsigned long long)blockIdx.x * blockDim.x + threadIdx.x; // t fastest
    const int t = (int)(i & 2047u);
    const unsigned long long r = i >> 11;   // (b*16 + h)*64 + d
    const int d  = (int)(r & 63u);
    const unsigned long long bh = r >> 6;
    const int hh = (int)(bh & 15u);
    const int bb = (int)(bh >> 4);
    vT[i] = qkv[((unsigned long long)(bb * 2048 + t)) * 3072ull + 2048ull +
                (unsigned long long)(hh * 64 + d)];
}

// ---------------------------------------------------------------------------
// GEMM: out[M,N] = A[M,K](bf16,row-major) * W[N,K](bf16,row-major)^T
// One wave computes a 32x64 tile: 2 A-tiles share every B-operand
// (8 WMMAs per 12 x b128 loads per K-step of 32).
// ---------------------------------------------------------------------------
template <bool STORE_F32>
__global__ __launch_bounds__(128)
void gemm_bf16_kernel(const __bf16* __restrict__ A,
                      const __bf16* __restrict__ W,
                      void* __restrict__ out, int M, int N, int K) {
    const int lane = threadIdx.x & 31;
    const int wave = threadIdx.x >> 5;
    const int row  = lane & 15;
    const int hl   = lane >> 4;
    const int tilesN = N >> 6;
    const int gid = blockIdx.x * 4 + wave;
    const int mT = gid / tilesN;
    const int nT = gid - mT * tilesN;
    const int m0 = mT * 32;
    if (m0 >= M) return;
    const int n0 = nT * 64;

    v8f acc[2][4];
#pragma unroll
    for (int i = 0; i < 2; ++i)
#pragma unroll
        for (int j = 0; j < 4; ++j)
            acc[i][j] = (v8f){0.f, 0.f, 0.f, 0.f, 0.f, 0.f, 0.f, 0.f};

    const int ka = hl * 8;    // A per-lane K base: {0..7,16..23}/{8..15,24..31}
    const int kb = hl * 8;    // B per-lane K base: {0..7,16..23}/{8..15,24..31}
    const __bf16* arow0 = A + (size_t)(m0 + row) * K;
    const __bf16* arow1 = arow0 + (size_t)16 * K;

    for (int kk = 0; kk < K; kk += 32) {
        V16 a0, a1;
        a0.h[0] = *(const v8bf*)(arow0 + kk + ka);
        a0.h[1] = *(const v8bf*)(arow0 + kk + ka + 16);
        a1.h[0] = *(const v8bf*)(arow1 + kk + ka);
        a1.h[1] = *(const v8bf*)(arow1 + kk + ka + 16);
#pragma unroll
        for (int j = 0; j < 4; ++j) {
            const __bf16* wrow = W + (size_t)(n0 + j * 16 + row) * K + kk + kb;
            V16 b;
            b.h[0] = *(const v8bf*)(wrow);
            b.h[1] = *(const v8bf*)(wrow + 16);
            acc[0][j] = wmma_bf16(a0.v, b.v, acc[0][j]);
            acc[1][j] = wmma_bf16(a1.v, b.v, acc[1][j]);
        }
    }

#pragma unroll
    for (int i = 0; i < 2; ++i)
#pragma unroll
        for (int j = 0; j < 4; ++j)
#pragma unroll
            for (int r = 0; r < 8; ++r) {
                const int orow = m0 + i * 16 + r + hl * 8;
                const int ocol = n0 + j * 16 + row;
                const float vv = acc[i][j][r];
                if (STORE_F32)
                    ((float*)out)[(size_t)orow * N + ocol] = vv;
                else
                    ((unsigned short*)out)[(size_t)orow * N + ocol] = f2bf(vv);
            }
}

// ---------------------------------------------------------------------------
// Flash attention: one wave per (b, h, 16-row q tile). dh=64, T=2048, H=16.
// K/V tiles (64 seq positions per step) staged into LDS via the Tensor Data
// Mover, double-buffered with s_wait_tensorcnt. 16 WMMAs per S-step of 64.
// ---------------------------------------------------------------------------
__global__ __launch_bounds__(32)
void attn_kernel(const __bf16* __restrict__ qkv,
                 const __bf16* __restrict__ vT,
                 unsigned short* __restrict__ ob) {
    const int qt = blockIdx.x;   // 0..127
    const int h  = blockIdx.y;   // 0..15
    const int b  = blockIdx.z;   // 0..1
    const int lane = threadIdx.x & 31;
    const int row  = lane & 15;
    const int hl   = lane >> 4;

    const int T = 2048, D3 = 3072, NH = 16;
    const int t0 = qt * 16;
    const int ka = hl * 8;
    const int kb = hl * 8;
    const float scale = 0.125f;  // 1/sqrt(64)

    __shared__ unsigned short kls[2][64 * 64];  // K tiles  [s][dh]   (2 x 8 KB)
    __shared__ unsigned short vls[2][64 * 64];  // V tiles  [dh][s]   (2 x 8 KB)
    __shared__ unsigned short pls[16 * 64];     // P bounce buffer    (2 KB)

    // Q tile: 16x64 as two A-operands (k = 0..31, 32..63)
    const __bf16* qrow = qkv + (size_t)(b * T + t0 + row) * D3 + h * 64;
    V16 aq[2];
#pragma unroll
    for (int kk = 0; kk < 2; ++kk) {
        aq[kk].h[0] = *(const v8bf*)(qrow + kk * 32 + ka);
        aq[kk].h[1] = *(const v8bf*)(qrow + kk * 32 + ka + 16);
    }

    const unsigned short* kbase =
        (const unsigned short*)qkv + (size_t)(b * T) * D3 + 1024 + h * 64;
    const unsigned short* vtb =
        (const unsigned short*)vT + (size_t)(b * NH + h) * 64 * T;

    v8f oacc[4];
#pragma unroll
    for (int j = 0; j < 4; ++j)
        oacc[j] = (v8f){0.f, 0.f, 0.f, 0.f, 0.f, 0.f, 0.f, 0.f};

    float mrun[8], lrun[8];
#pragma unroll
    for (int r = 0; r < 8; ++r) { mrun[r] = -3.0e38f; lrun[r] = 0.f; }

    auto issue = [&](int buf, int s0) {
        __syncthreads();  // drain pending LDS reads before DMA overwrites
        tile_load_64x64(&kls[buf][0], kbase + (size_t)s0 * D3, D3, lane);
        tile_load_64x64(&vls[buf][0], vtb + s0, T, lane);
    };

    auto compute = [&](int buf) {
        // ---- scores: four 16x16 tiles (64 S-columns) ----
        v8f sc[4];
#pragma unroll
        for (int j = 0; j < 4; ++j)
            sc[j] = (v8f){0.f, 0.f, 0.f, 0.f, 0.f, 0.f, 0.f, 0.f};
#pragma unroll
        for (int kk = 0; kk < 2; ++kk) {        // dh halves
#pragma unroll
            for (int j = 0; j < 4; ++j) {       // s sub-tile
                const unsigned short* kr =
                    &kls[buf][(j * 16 + row) * 64 + kk * 32 + kb];
                V16 bt;
                bt.h[0] = *(const v8bf*)(const void*)(kr);
                bt.h[1] = *(const v8bf*)(const void*)(kr + 16);
                sc[j] = wmma_bf16(aq[kk].v, bt.v, sc[j]);
            }
        }

        // ---- online softmax update over 64 columns ----
        float al[8];
#pragma unroll
        for (int r = 0; r < 8; ++r) {
            float x0 = sc[0][r] * scale;
            float x1 = sc[1][r] * scale;
            float x2 = sc[2][r] * scale;
            float x3 = sc[3][r] * scale;
            float mrow = rowmax16(fmaxf(fmaxf(x0, x1), fmaxf(x2, x3)));
            float nm = fmaxf(mrun[r], mrow);
            float alpha = __expf(mrun[r] - nm);
            mrun[r] = nm;
            float p0 = __expf(x0 - nm);
            float p1 = __expf(x1 - nm);
            float p2 = __expf(x2 - nm);
            float p3 = __expf(x3 - nm);
            sc[0][r] = p0; sc[1][r] = p1; sc[2][r] = p2; sc[3][r] = p3;
            lrun[r] = lrun[r] * alpha + rowsum16((p0 + p1) + (p2 + p3));
            al[r] = alpha;
        }
#pragma unroll
        for (int j = 0; j < 4; ++j)
#pragma unroll
            for (int r = 0; r < 8; ++r) oacc[j][r] *= al[r];

        // ---- P (C-layout f32) -> LDS -> two A-layout bf16 operands ----
#pragma unroll
        for (int j = 0; j < 4; ++j)
#pragma unroll
            for (int r = 0; r < 8; ++r)
                pls[(r + hl * 8) * 64 + j * 16 + row] = f2bf(sc[j][r]);
        __syncthreads();
        V16 pa[2];
#pragma unroll
        for (int kk = 0; kk < 2; ++kk) {
            const unsigned short* pr = &pls[row * 64 + kk * 32 + ka];
            pa[kk].h[0] = *(const v8bf*)(const void*)(pr);
            pa[kk].h[1] = *(const v8bf*)(const void*)(pr + 16);
        }
        __syncthreads();

        // ---- out += P @ V : 4 dh-tiles x 2 K-steps = 8 WMMAs ----
#pragma unroll
        for (int j = 0; j < 4; ++j)
#pragma unroll
            for (int kk = 0; kk < 2; ++kk) {
                const unsigned short* vr =
                    &vls[buf][(j * 16 + row) * 64 + kk * 32 + kb];
                V16 bt;
                bt.h[0] = *(const v8bf*)(const void*)(vr);
                bt.h[1] = *(const v8bf*)(const void*)(vr + 16);
                oacc[j] = wmma_bf16(pa[kk].v, bt.v, oacc[j]);
            }
    };

    // ---- double-buffered TDM pipeline over S (T/64 = 32 steps) ----
    issue(0, 0);
    for (int s0 = 0; s0 < T; s0 += 128) {
        issue(1, s0 + 64);          // always valid: T % 128 == 0
        tdm_wait_le2();             // buffer 0 tiles landed
        compute(0);
        if (s0 + 128 < T) {
            issue(0, s0 + 128);
            tdm_wait_le2();         // buffer 1 tiles landed
        } else {
            tdm_wait_0();
        }
        compute(1);
    }

    // ---- normalize and store ----
#pragma unroll
    for (int r = 0; r < 8; ++r) lrun[r] = 1.0f / lrun[r];
#pragma unroll
    for (int j = 0; j < 4; ++j)
#pragma unroll
        for (int r = 0; r < 8; ++r) {
            const int orow = t0 + r + hl * 8;
            const int ocol = h * 64 + j * 16 + row;
            ob[(size_t)(b * T + orow) * 1024 + ocol] =
                f2bf(oacc[j][r] * lrun[r]);
        }
}

// ---------------------------------------------------------------------------
// Host launcher
// ---------------------------------------------------------------------------
extern "C" void kernel_launch(void* const* d_in, const int* in_sizes, int n_in,
                              void* d_out, int out_size, void* d_ws,
                              size_t ws_size, hipStream_t stream) {
    const float* x     = (const float*)d_in[0];
    const float* wqkv  = (const float*)d_in[1];
    const float* wproj = (const float*)d_in[2];
    float* out = (float*)d_out;

    const size_t BT = 4096, D = 1024, D3 = 3072, T = 2048;
    unsigned short* xb     = (unsigned short*)d_ws;        //  8 MB
    unsigned short* wqkvb  = xb + BT * D;                   //  6 MB
    unsigned short* wprojb = wqkvb + D3 * D;                //  2 MB
    unsigned short* qkvb   = wprojb + D * D;                // 24 MB
    unsigned short* vTb    = qkvb + BT * D3;                //  8 MB
    unsigned short* obb    = vTb + (size_t)2 * 16 * 64 * T; //  8 MB

    // 1) convert inputs to bf16
    f32_to_bf16_kernel<<<(unsigned)((BT * D + 255) / 256), 256, 0, stream>>>(
        x, xb, BT * D);
    f32_to_bf16_kernel<<<(unsigned)((D3 * D + 255) / 256), 256, 0, stream>>>(
        wqkv, wqkvb, D3 * D);
    f32_to_bf16_kernel<<<(unsigned)((D * D + 255) / 256), 256, 0, stream>>>(
        wproj, wprojb, D * D);

    // 2) QKV GEMM: [4096,3072] = x[4096,1024] @ w_qkv[3072,1024]^T (bf16 out)
    gemm_bf16_kernel<false><<<(128 * 48) / 4, 128, 0, stream>>>(
        (const __bf16*)xb, (const __bf16*)wqkvb, qkvb, 4096, 3072, 1024);

    // 3) transpose V -> vT[b,h,dh,T]
    transpose_v_kernel<<<(unsigned)((2ull * 16 * 64 * T) / 256), 256, 0,
                         stream>>>(qkvb, vTb);

    // 4) flash attention (TDM-staged K/V tiles)
    dim3 ag(128, 16, 2);
    attn_kernel<<<ag, 32, 0, stream>>>((const __bf16*)qkvb,
                                       (const __bf16*)vTb, obb);

    // 5) output projection: out[4096,1024] = ob @ w_proj^T (f32 out)
    gemm_bf16_kernel<true><<<(128 * 16) / 4, 128, 0, stream>>>(
        (const __bf16*)obb, (const __bf16*)wprojb, out, 4096, 1024, 1024);
}
